// LSTMModel_75960791597987
// MI455X (gfx1250) — compile-verified, matrix operand order checked by reading, not run on hardware
//
#include <hip/hip_runtime.h>

#define I_DIM 128
#define H_DIM 512
#define FC_DIM 128
#define B_DIM 512
#define T_DIM 256

typedef __attribute__((ext_vector_type(16))) __bf16 v16bf;
typedef __attribute__((ext_vector_type(8)))  __bf16 v8bf;
typedef __attribute__((ext_vector_type(4)))  __bf16 v4bf;
typedef __attribute__((ext_vector_type(8)))  float  v8f;

// Load one 16x32 bf16 tile (row-major, leading dim ld) from GLOBAL into the
// WMMA A/B register layout for v_wmma_f32_16x16x32_bf16 (wave32):
//   lane l: row = l&15, khalf = l>>4
//   elems 0..7  = K[khalf*8 + 0..7]
//   elems 8..15 = K[16 + khalf*8 + 0..7]
__device__ __forceinline__ v16bf load_tile_bf16(const __bf16* __restrict__ base,
                                                int ld, int k0, int lane) {
  const int rr = lane & 15;
  const int kg = (lane >> 4) & 1;
  const __bf16* q = base + (size_t)rr * (size_t)ld + (size_t)(k0 + kg * 8);
  v8bf lo = *(const v8bf*)(q);
  v8bf hi = *(const v8bf*)(q + 16);
  v16bf t;
#pragma unroll
  for (int e = 0; e < 8; ++e) { t[e] = lo[e]; t[e + 8] = hi[e]; }
  return t;
}

// One LSTM timestep. Block = 256 threads = 8 waves, all sharing one hidden
// tile j0. Each wave owns one 16-row batch tile and 4 gate accumulators.
// Weight tiles (4 gates x 16x32 bf16 = 4KB per K-step) are staged into
// double-buffered LDS with async global->LDS copies (ASYNCcnt) and consumed
// via ds_load_b128 by all 8 waves. The K-loop is fully unrolled; A tiles are
// register-prefetched one K-step ahead; all 8 DS loads of a K-step are issued
// before the first WMMA so dscnt waits are partial and overlap WMMA issue.
__global__ __launch_bounds__(256) void lstm_step_kernel(
    const __bf16* __restrict__ xb,      // [B, T, I] bf16
    const __bf16* __restrict__ Wih,     // [4H, I]   bf16
    const __bf16* __restrict__ Whh,     // [4H, H]   bf16
    const float*  __restrict__ b_ih,    // [4H]
    const float*  __restrict__ b_hh,    // [4H]
    const __bf16* __restrict__ h_prev,  // [B, H]    bf16
    __bf16* __restrict__ h_next,        // [B, H]    bf16
    float*  __restrict__ c_state,       // [B, H]    f32 (in/out)
    float*  __restrict__ h_out_f32,     // [B, H]    f32 (written when is_last)
    int t, int is_last)
{
  __shared__ __align__(16) char smem[2 * 4096];  // double-buffered B tiles

  const int tid  = threadIdx.x;
  const int lane = tid & 31;
  const int wave = tid >> 5;                     // 8 waves/block
  const int j0   = blockIdx.x * 16;              // hidden tile (shared)
  const int m0   = (blockIdx.y * 8 + wave) * 16; // batch tile (per wave)

  // --- staging coordinates: 64 rows (4 gates x 16) of 64B; 16B per thread ---
  const int srow = tid >> 2;                     // 0..63
  const int sseg = tid & 3;                      // 16B segment within row
  const int sg   = srow >> 4;                    // gate
  const int sr   = srow & 15;                    // row within gate tile
  const __bf16* wsrc_ih = Wih + (size_t)(sg * H_DIM + j0 + sr) * I_DIM + sseg * 8;
  const __bf16* wsrc_hh = Whh + (size_t)(sg * H_DIM + j0 + sr) * H_DIM + sseg * 8;
  // 32-bit LDS byte offsets of this thread's staging slot in each buffer
  const unsigned lds_slot0 =
      (unsigned)(uintptr_t)(&smem[0]) + (unsigned)(srow * 64 + sseg * 16);
  const unsigned lds_slot1 = lds_slot0 + 4096u;

  constexpr int NK = (I_DIM + H_DIM) / 32;       // 20 K-steps total

  auto stage = [&](int kk, int buf) {
    const __bf16* src = (kk < 4) ? (wsrc_ih + kk * 32) : (wsrc_hh + (kk - 4) * 32);
    unsigned dst = buf ? lds_slot1 : lds_slot0;
    asm volatile("global_load_async_to_lds_b128 %0, %1, off"
                 :: "v"(dst), "v"((unsigned long long)(uintptr_t)src)
                 : "memory");
  };

  v8f zero = {};
  v8f acc[4];
#pragma unroll
  for (int g = 0; g < 4; ++g) acc[g] = zero;

  const __bf16* xA = xb + (size_t)m0 * (size_t)(T_DIM * I_DIM) + (size_t)t * I_DIM;
  const __bf16* hA = h_prev + (size_t)m0 * H_DIM;
  const int n  = lane & 15;
  const int kg = (lane >> 4) & 1;

  stage(0, 0);                                   // prime async pipeline
  v16bf a_cur = load_tile_bf16(xA, T_DIM * I_DIM, 0, lane);  // prime A regs

#pragma unroll
  for (int kk = 0; kk < NK; ++kk) {
    const int cur = kk & 1;
    if (kk + 1 < NK) {
      stage(kk + 1, cur ^ 1);                    // prefetch next B tiles -> LDS
      asm volatile("s_wait_asynccnt 0x1" ::: "memory");  // current buf arrived
    } else {
      asm volatile("s_wait_asynccnt 0x0" ::: "memory");
    }
    __syncthreads();                             // all waves' pieces visible

    // Register-prefetch next A tile (x_t rows for kk<4, h_prev rows after);
    // it has the whole current K-step to land.
    v16bf a_next = a_cur;
    if (kk + 1 < NK) {
      a_next = (kk + 1 < 4)
                   ? load_tile_bf16(xA, T_DIM * I_DIM, (kk + 1) * 32, lane)
                   : load_tile_bf16(hA, H_DIM, (kk + 1 - 4) * 32, lane);
    }

    // Issue ALL 8 ds_load_b128 for the 4 gate tiles first...
    const char* bb = &smem[cur * 4096];
    v8bf blo[4], bhi[4];
#pragma unroll
    for (int g = 0; g < 4; ++g) {
      const char* bp = bb + (size_t)((g * 16 + n) * 64 + kg * 16);
      blo[g] = *(const v8bf*)(bp);
      bhi[g] = *(const v8bf*)(bp + 32);
    }
    // ...then 4 back-to-back WMMAs (distinct accumulators -> no D->A/B hazard)
#pragma unroll
    for (int g = 0; g < 4; ++g) {
      v16bf b;
#pragma unroll
      for (int e = 0; e < 8; ++e) { b[e] = blo[g][e]; b[e + 8] = bhi[g][e]; }
      acc[g] = __builtin_amdgcn_wmma_f32_16x16x32_bf16(
          false, a_cur, false, b, (short)0, acc[g], false, false);
    }
    __syncthreads();                             // close WAR before next prefetch
    a_cur = a_next;
  }

  // ---- elementwise: C/D layout -> lane holds column n, rows m = r + 8*(lane>>4) ----
  const int half = lane >> 4;
  const int j    = j0 + n;

  float bias[4];
#pragma unroll
  for (int g = 0; g < 4; ++g) bias[g] = b_ih[g * H_DIM + j] + b_hh[g * H_DIM + j];

#pragma unroll
  for (int r = 0; r < 8; ++r) {
    const int m = m0 + r + 8 * half;
    const size_t idx = (size_t)m * H_DIM + j;
    const float ig = 1.0f / (1.0f + __expf(-(acc[0][r] + bias[0])));
    const float fg = 1.0f / (1.0f + __expf(-(acc[1][r] + bias[1])));
    const float gg = tanhf(acc[2][r] + bias[2]);
    const float og = 1.0f / (1.0f + __expf(-(acc[3][r] + bias[3])));
    const float c  = fg * c_state[idx] + ig * gg;
    c_state[idx] = c;
    const float h = og * tanhf(c);
    h_next[idx] = (__bf16)h;
    if (is_last) h_out_f32[idx] = h;
  }
}

// fp32 -> bf16 conversion of x / W_ih / W_hh, and zero h0 + c state.
__global__ void convert_init_kernel(
    const float* __restrict__ x, const float* __restrict__ Wih, const float* __restrict__ Whh,
    __bf16* __restrict__ xb, __bf16* __restrict__ Wihb, __bf16* __restrict__ Whhb,
    __bf16* __restrict__ h0, float* __restrict__ c0,
    int nx4, int nwih, int nwhh, int nh)
{
  const int idx = blockIdx.x * blockDim.x + threadIdx.x;
  const int stride = gridDim.x * blockDim.x;
  const float4* x4 = (const float4*)x;
  v4bf* xb4 = (v4bf*)xb;
  for (int i = idx; i < nx4; i += stride) {
    float4 v = x4[i];
    v4bf o;
    o[0] = (__bf16)v.x; o[1] = (__bf16)v.y; o[2] = (__bf16)v.z; o[3] = (__bf16)v.w;
    xb4[i] = o;
  }
  for (int i = idx; i < nwih; i += stride) Wihb[i] = (__bf16)Wih[i];
  for (int i = idx; i < nwhh; i += stride) Whhb[i] = (__bf16)Whh[i];
  for (int i = idx; i < nh;   i += stride) { h0[i] = (__bf16)0.0f; c0[i] = 0.0f; }
}

// out1 = relu(h_n @ W1^T + b1)   [B, FC]
__global__ void fc1_kernel(const float* __restrict__ h, const float* __restrict__ W1,
                           const float* __restrict__ b1, float* __restrict__ out) {
  const int idx = blockIdx.x * blockDim.x + threadIdx.x;
  if (idx >= B_DIM * FC_DIM) return;
  const int b = idx / FC_DIM, n = idx % FC_DIM;
  const float* hr = h  + (size_t)b * H_DIM;
  const float* wr = W1 + (size_t)n * H_DIM;
  float s = b1[n];
#pragma unroll 8
  for (int k = 0; k < H_DIM; ++k) s = fmaf(hr[k], wr[k], s);
  out[idx] = fmaxf(s, 0.0f);
}

// out[b] = fc1[b,:] @ W2 + b2
__global__ void fc2_kernel(const float* __restrict__ f1, const float* __restrict__ W2,
                           const float* __restrict__ b2, float* __restrict__ out) {
  const int b = blockIdx.x * blockDim.x + threadIdx.x;
  if (b >= B_DIM) return;
  float s = b2[0];
#pragma unroll 8
  for (int k = 0; k < FC_DIM; ++k) s = fmaf(f1[(size_t)b * FC_DIM + k], W2[k], s);
  out[b] = s;
}

extern "C" void kernel_launch(void* const* d_in, const int* in_sizes, int n_in,
                              void* d_out, int out_size, void* d_ws, size_t ws_size,
                              hipStream_t stream) {
  const float* x    = (const float*)d_in[0];
  const float* Wih  = (const float*)d_in[1];
  const float* Whh  = (const float*)d_in[2];
  const float* bih  = (const float*)d_in[3];
  const float* bhh  = (const float*)d_in[4];
  const float* W1   = (const float*)d_in[5];
  const float* b1   = (const float*)d_in[6];
  const float* W2   = (const float*)d_in[7];
  const float* b2   = (const float*)d_in[8];
  float* out = (float*)d_out;

  char* ws = (char*)d_ws;
  size_t off = 0;
  auto alloc = [&](size_t bytes) -> void* {
    off = (off + 255) & ~(size_t)255;
    void* p = ws + off;
    off += bytes;
    return p;
  };

  const int nx   = B_DIM * T_DIM * I_DIM;
  const int nwih = 4 * H_DIM * I_DIM;
  const int nwhh = 4 * H_DIM * H_DIM;
  const int nh   = B_DIM * H_DIM;

  __bf16* xb     = (__bf16*)alloc((size_t)nx * 2);
  __bf16* Wihb   = (__bf16*)alloc((size_t)nwih * 2);
  __bf16* Whhb   = (__bf16*)alloc((size_t)nwhh * 2);
  __bf16* hping0 = (__bf16*)alloc((size_t)nh * 2);
  __bf16* hping1 = (__bf16*)alloc((size_t)nh * 2);
  float*  cst    = (float*)alloc((size_t)nh * 4);
  float*  hf32   = (float*)alloc((size_t)nh * 4);
  float*  f1buf  = (float*)alloc((size_t)B_DIM * FC_DIM * 4);

  // 1) bf16 conversion + state init (re-zeros h0/c every call -> deterministic)
  convert_init_kernel<<<2048, 256, 0, stream>>>(x, Wih, Whh, xb, Wihb, Whhb,
                                                hping0, cst, nx / 4, nwih, nwhh, nh);

  // 2) sequential LSTM scan, ping-pong h buffers
  __bf16* hp[2] = {hping0, hping1};
  dim3 grid(H_DIM / 16, B_DIM / (8 * 16));  // 32 x 4 blocks, 256 threads
  for (int t = 0; t < T_DIM; ++t) {
    lstm_step_kernel<<<grid, 256, 0, stream>>>(
        xb, Wihb, Whhb, bih, bhh,
        hp[t & 1], hp[(t + 1) & 1], cst, hf32,
        t, (t == T_DIM - 1) ? 1 : 0);
  }

  // 3) FC head
  fc1_kernel<<<(B_DIM * FC_DIM + 255) / 256, 256, 0, stream>>>(hf32, W1, b1, f1buf);
  fc2_kernel<<<(B_DIM + 255) / 256, 256, 0, stream>>>(f1buf, W2, b2, out);
}
